// MTGNNLayer_29764123361448
// MI455X (gfx1250) — matrix operand order repaired and checked by reading, not compiled, not run on hardware
//
#include <hip/hip_runtime.h>

// ---------------- types / helpers ----------------
typedef __attribute__((ext_vector_type(16))) __bf16 v16bf;
typedef __attribute__((ext_vector_type(8)))  float  v8f;
typedef __attribute__((ext_vector_type(4)))  unsigned int u32x4;
typedef __attribute__((ext_vector_type(4)))  int i32x4;
typedef __attribute__((ext_vector_type(8)))  int i32x8;

#if defined(__has_builtin)
#  if __has_builtin(__builtin_amdgcn_tensor_load_to_lds)
#    define HAVE_TDM 1
#  else
#    define HAVE_TDM 0
#  endif
#else
#  define HAVE_TDM 0
#endif

__device__ __forceinline__ unsigned short f2bf(float f) {
  unsigned int u = __float_as_uint(f);
  u += 0x7FFFu + ((u >> 16) & 1u);          // round-to-nearest-even
  return (unsigned short)(u >> 16);
}
__device__ __forceinline__ float bf2f(unsigned short s) {
  return __uint_as_float(((unsigned int)s) << 16);
}

union FragBF { v16bf v; uint4 q[2]; unsigned short s[16]; };
union Pack8  { unsigned short s[8]; uint4 q; };

// ---------------- problem constants ----------------
#define Bb   32
#define CRc  32
#define CCc  32
#define CSc  64
#define Nn   500
#define Tt   64
#define Kk   52
#define NPAD 512
#define COLS 53248        // B*CC*K
#define KSKIP 1664        // CC*K
#define SCOLS 16000       // B*N
#define ALPHA 0.05f

constexpr size_t XN_ELEMS = (size_t)Bb*CRc*Nn*Kk;     // 26,624,000

// workspace layout (bytes)
constexpr size_t SZ_A   = (size_t)NPAD*NPAD*2;
constexpr size_t SZ_H   = (size_t)COLS*NPAD*2;
constexpr size_t SZ_HS  = (size_t)SCOLS*KSKIP*2;
constexpr size_t SZ_WSK = (size_t)CSc*KSKIP*2;
constexpr size_t OFF_A1   = 0;
constexpr size_t OFF_A2   = OFF_A1 + SZ_A;
constexpr size_t OFF_H0   = OFF_A2 + SZ_A;
constexpr size_t OFF_H1A  = OFF_H0  + SZ_H;
constexpr size_t OFF_H2A  = OFF_H1A + SZ_H;
constexpr size_t OFF_H1B  = OFF_H2A + SZ_H;
constexpr size_t OFF_H2B  = OFF_H1B + SZ_H;
constexpr size_t OFF_HS   = OFF_H2B + SZ_H;
constexpr size_t OFF_WSK  = OFF_HS  + SZ_HS;
constexpr size_t OFF_PART = OFF_WSK + SZ_WSK;
constexpr size_t OFF_SUMS = OFF_PART + (size_t)6656*2*4;

// ---------------- TDM 2-D tile load: global(bf16, row stride) -> LDS (80B padded rows) ----------
// D# per CDNA5 ISA ch.8: group0 = count/lds_addr/global_addr/type, group1 = sizes+strides+pad.
// pad_interval code 3 => every 16 DWORDs (one 64B tile row), pad_amount code 3 => 4 DWORDs (16B)
// so LDS row stride becomes 80B, matching the [rows][40]-ushort LDS arrays (conflict-free b128).
__device__ __forceinline__ void tdm_load_tile(const unsigned short* gsrc,
                                              unsigned lds_byte_off,
                                              int tile_k_elems, int tile_rows,
                                              int row_stride_elems) {
#if HAVE_TDM
  unsigned long long ga = (unsigned long long)gsrc;
  u32x4 g0;
  g0[0] = 1u;                                              // count=1 (valid user D#)
  g0[1] = lds_byte_off;                                    // lds_addr
  g0[2] = (unsigned)ga;                                    // global_addr[31:0]
  g0[3] = (unsigned)((ga >> 32) & 0x1FFFFFFull) | (2u << 30); // global_addr[56:32], type=2
  i32x8 g1;
  const unsigned td0 = 1u << 20, td1 = 1u << 20;           // generous dims: tiles never OOB
  g1[0] = (int)((1u << 16) | (1u << 20) | (3u << 22) | (3u << 25)); // data_size=2B, pad en, int=16dw, amt=4dw
  g1[1] = (int)((td0 & 0xFFFFu) << 16);                    // tensor_dim0 lo16 (bits 63:48)
  g1[2] = (int)(((td0 >> 16) & 0xFFFFu) | ((td1 & 0xFFFFu) << 16));  // dim0 hi / dim1 lo
  g1[3] = (int)(((td1 >> 16) & 0xFFFFu) | ((unsigned)tile_k_elems << 16)); // dim1 hi / tile_dim0
  g1[4] = (int)((unsigned)tile_rows & 0xFFFFu);            // tile_dim1 (tile_dim2=0)
  g1[5] = (int)(unsigned)row_stride_elems;                 // tensor_dim0_stride[31:0]
  g1[6] = 0;                                               // stride hi / dim1_stride lo
  g1[7] = 0;
  i32x4 z4 = {0, 0, 0, 0};
#if __clang_major__ >= 23
  i32x8 z8 = {0, 0, 0, 0, 0, 0, 0, 0};
  __builtin_amdgcn_tensor_load_to_lds(g0, g1, z4, z4, z8, 0);
#else
  __builtin_amdgcn_tensor_load_to_lds(g0, g1, z4, z4, 0);
#endif
#else
  (void)gsrc; (void)lds_byte_off; (void)tile_k_elems; (void)tile_rows; (void)row_stride_elems;
#endif
}

// ---------------- adjacency prep: (adp+I)/rowsum and (adp^T+I)/rowsum, bf16, 512x512 zero-padded
__global__ __launch_bounds__(256) void k_adj(const float* __restrict__ adp,
                                             unsigned short* __restrict__ A1,
                                             unsigned short* __restrict__ A2) {
  int v = blockIdx.x * 256 + threadIdx.x;   // 0..1023
  if (v < NPAD) {
    unsigned short* row = A1 + (size_t)v * NPAD;
    if (v < Nn) {
      float s = 1.f;
      for (int w = 0; w < Nn; ++w) s += adp[v * Nn + w];
      float inv = 1.f / s;
      for (int w = 0; w < Nn; ++w)
        row[w] = f2bf((adp[v * Nn + w] + (w == v ? 1.f : 0.f)) * inv);
      for (int w = Nn; w < NPAD; ++w) row[w] = 0;
    } else {
      for (int w = 0; w < NPAD; ++w) row[w] = 0;
    }
  } else {
    int v2 = v - NPAD;
    unsigned short* row = A2 + (size_t)v2 * NPAD;
    if (v2 < Nn) {
      float s = 1.f;
      for (int w = 0; w < Nn; ++w) s += adp[w * Nn + v2];
      float inv = 1.f / s;
      for (int w = 0; w < Nn; ++w)
        row[w] = f2bf((adp[w * Nn + v2] + (w == v2 ? 1.f : 0.f)) * inv);
      for (int w = Nn; w < NPAD; ++w) row[w] = 0;
    } else {
      for (int w = 0; w < NPAD; ++w) row[w] = 0;
    }
  }
}

// ---------------- f32 -> bf16 convert (skip weights)
__global__ __launch_bounds__(256) void k_cvt(const float* __restrict__ src,
                                             unsigned short* __restrict__ dst, int nel) {
  int i = blockIdx.x * 256 + threadIdx.x;
  if (i < nel) dst[i] = f2bf(src[i]);
}

// ---------------- zero node-padding of H0 (cols x nodes 500..511)
__global__ __launch_bounds__(256) void k_padH(unsigned short* __restrict__ H0) {
  int i = blockIdx.x * 256 + threadIdx.x;     // 53248*12 = 638976 threads
  int col = i / 12, p = i % 12;
  H0[(size_t)col * NPAD + Nn + p] = 0;
}

// ---------------- dilated inception + tanh*sigmoid gating ----------------
struct GateArgs {
  const float* x;
  const float* fw[4]; const float* fb[4];
  const float* gw[4]; const float* gb[4];
  unsigned short* H0;     // [col=(b*32+c)*52+l][node]
  unsigned short* Hskip;  // [(b*500+n)][c*52+l]
};

__global__ __launch_bounds__(256) void k_gate(GateArgs a) {
  __shared__ float xs[CCc][Tt];     // x[b,:,n,:]
  __shared__ float wls[9216];       // f weights [0..4607], g weights [4608..]
  __shared__ float bls[64];         // f biases [0..31], g biases [32..63]
  const int ks[4]   = {2, 3, 6, 7};
  const int foff[4] = {0, 512, 1280, 2816};

  int b = blockIdx.x / Nn, n = blockIdx.x % Nn;
  int tid = threadIdx.x;

  for (int i = 0; i < 8; ++i) {
    int idx = tid + i * 256;
    int ci = idx >> 6, t = idx & 63;
    xs[ci][t] = a.x[(((size_t)b * CRc + ci) * Nn + n) * Tt + t];
  }
  for (int kb = 0; kb < 4; ++kb) {
    int cnt = 8 * 32 * ks[kb];
    for (int i = tid; i < cnt; i += 256) {
      wls[foff[kb] + i]        = a.fw[kb][i];
      wls[4608 + foff[kb] + i] = a.gw[kb][i];
    }
  }
  if (tid < 64) {
    int gate = tid >> 5, rem = tid & 31, kb = rem >> 3, cw = rem & 7;
    bls[tid] = gate ? a.gb[kb][cw] : a.fb[kb][cw];
  }
  __syncthreads();

  for (int j = tid; j < CCc * Kk; j += 256) {
    int c = j / Kk, l = j % Kk;
    int kb = c >> 3, cw = c & 7;
    int kk = ks[kb];
    int base = l + 12 - 2 * (kk - 1);
    float accf = bls[kb * 8 + cw];
    float accg = bls[32 + kb * 8 + cw];
    const float* wf = &wls[foff[kb] + cw * 32 * kk];
    const float* wg = wf + 4608;
    for (int ci = 0; ci < 32; ++ci) {
      const float* xr  = &xs[ci][base];
      const float* wfc = wf + ci * kk;
      const float* wgc = wg + ci * kk;
      for (int t = 0; t < kk; ++t) {
        float xv = xr[2 * t];
        accf += xv * wfc[t];
        accg += xv * wgc[t];
      }
    }
    float h = tanhf(accf) * (1.f / (1.f + expf(-accg)));
    unsigned short hb = f2bf(h);
    a.H0[((size_t)(b * CCc + c) * Kk + l) * NPAD + n] = hb;
    a.Hskip[((size_t)(b * Nn + n)) * KSKIP + c * Kk + l] = hb;
  }
}

// ---------------- mixprop propagation GEMM: Hout[col][v] = a*H0[col][v] + .95 * sum_w A[v][w]*Hin[col][w]
// TDM double-buffered LDS tiles; wave0 drives the DMA, 8 waves do 2x2 WMMA tiles each.
__global__ __launch_bounds__(256) void k_gemm_prop(const unsigned short* __restrict__ A,
                                                   const unsigned short* __restrict__ Hin,
                                                   const unsigned short* __restrict__ H0,
                                                   unsigned short* __restrict__ Hout) {
  __shared__ __align__(16) unsigned short as[2][64][40];    // 80B rows
  __shared__ __align__(16) unsigned short hs[2][128][40];
  const int tid  = threadIdx.x;
  const int wave = tid >> 5, lane = tid & 31;
  const int n16  = lane & 15, half = lane >> 4;
  const int wm   = wave & 1, wc = wave >> 1;
  const int vBase   = blockIdx.x * 64;
  const int colBase = blockIdx.y * 128;

#if HAVE_TDM
  const unsigned asOff0 = (unsigned)(unsigned long long)(void*)&as[0][0][0];
  const unsigned asOff1 = (unsigned)(unsigned long long)(void*)&as[1][0][0];
  const unsigned hsOff0 = (unsigned)(unsigned long long)(void*)&hs[0][0][0];
  const unsigned hsOff1 = (unsigned)(unsigned long long)(void*)&hs[1][0][0];
  if (wave == 0) {
    tdm_load_tile(A   + (size_t)vBase   * NPAD, asOff0, 32, 64,  NPAD);
    tdm_load_tile(Hin + (size_t)colBase * NPAD, hsOff0, 32, 128, NPAD);
  }
#endif

  v8f acc[2][2] = {};
  int buf = 0;
  for (int k0 = 0; k0 < NPAD; k0 += 32) {
#if HAVE_TDM
    if (wave == 0) {
      if (k0 + 32 < NPAD) {
        tdm_load_tile(A   + (size_t)vBase   * NPAD + k0 + 32, buf ? asOff0 : asOff1, 32, 64,  NPAD);
        tdm_load_tile(Hin + (size_t)colBase * NPAD + k0 + 32, buf ? hsOff0 : hsOff1, 32, 128, NPAD);
        __builtin_amdgcn_s_wait_tensorcnt(2);   // current tile's 2 DMAs done
      } else {
        __builtin_amdgcn_s_wait_tensorcnt(0);
      }
    }
    __syncthreads();
#else
    { // manual staging fallback
      int r = tid >> 2, seg = tid & 3;
      *(uint4*)&as[0][r][seg * 8] = *(const uint4*)(A + (size_t)(vBase + r) * NPAD + k0 + seg * 8);
      int c = tid >> 1, sg = tid & 1;
      const uint4* src = (const uint4*)(Hin + (size_t)(colBase + c) * NPAD + k0 + sg * 16);
      *(uint4*)&hs[0][c][sg * 16]     = src[0];
      *(uint4*)&hs[0][c][sg * 16 + 8] = src[1];
      __syncthreads();
    }
#endif

    FragBF af[2], bf[2];
#pragma unroll
    for (int ms = 0; ms < 2; ++ms) {
      int rA = wm * 32 + ms * 16 + n16;
      af[ms].q[0] = *(const uint4*)&as[buf][rA][8 * half];        // K = 8h .. 8h+7
      af[ms].q[1] = *(const uint4*)&as[buf][rA][16 + 8 * half];   // K = 16+8h .. +7
    }
#pragma unroll
    for (int cs = 0; cs < 2; ++cs) {
      int cB = wc * 32 + cs * 16 + n16;
      bf[cs].q[0] = *(const uint4*)&hs[buf][cB][8 * half];        // K = 8h .. 8h+7
      bf[cs].q[1] = *(const uint4*)&hs[buf][cB][16 + 8 * half];   // K = 16+8h .. +7
    }
#pragma unroll
    for (int ms = 0; ms < 2; ++ms)
#pragma unroll
      for (int cs = 0; cs < 2; ++cs)
        acc[ms][cs] = __builtin_amdgcn_wmma_f32_16x16x32_bf16(
            false, af[ms].v, false, bf[cs].v, (short)0, acc[ms][cs], false, false);
    __syncthreads();
#if HAVE_TDM
    buf ^= 1;
#endif
  }

#pragma unroll
  for (int ms = 0; ms < 2; ++ms)
#pragma unroll
    for (int cs = 0; cs < 2; ++cs) {
      int col = colBase + wc * 32 + cs * 16 + n16;
      int v0  = vBase + wm * 32 + ms * 16 + 8 * half;   // 8 consecutive rows per lane
      size_t base = (size_t)col * NPAD + v0;
      const unsigned short* h0p = H0 + base;
      Pack8 op;
#pragma unroll
      for (int i = 0; i < 8; ++i)
        op.s[i] = f2bf(ALPHA * bf2f(h0p[i]) + (1.f - ALPHA) * acc[ms][cs][i]);
      *(uint4*)(Hout + base) = op.q;
    }
}

// ---------------- skip conv GEMM: s[b,cs,n] = bias[cs] + sum_j W[cs][j]*Hskip[(b,n)][j]
__global__ __launch_bounds__(256) void k_gemm_skip(const unsigned short* __restrict__ W,
                                                   const unsigned short* __restrict__ H,
                                                   const float* __restrict__ bias,
                                                   float* __restrict__ outS) {
  __shared__ __align__(16) unsigned short as[2][64][40];
  __shared__ __align__(16) unsigned short hs[2][128][40];
  const int tid  = threadIdx.x;
  const int wave = tid >> 5, lane = tid & 31;
  const int n16  = lane & 15, half = lane >> 4;
  const int wm   = wave & 1, wc = wave >> 1;
  const int colBase = blockIdx.x * 128;

#if HAVE_TDM
  const unsigned asOff0 = (unsigned)(unsigned long long)(void*)&as[0][0][0];
  const unsigned asOff1 = (unsigned)(unsigned long long)(void*)&as[1][0][0];
  const unsigned hsOff0 = (unsigned)(unsigned long long)(void*)&hs[0][0][0];
  const unsigned hsOff1 = (unsigned)(unsigned long long)(void*)&hs[1][0][0];
  if (wave == 0) {
    tdm_load_tile(W, asOff0, 32, 64, KSKIP);
    tdm_load_tile(H + (size_t)colBase * KSKIP, hsOff0, 32, 128, KSKIP);
  }
#endif

  v8f acc[2][2] = {};
  int buf = 0;
  for (int k0 = 0; k0 < KSKIP; k0 += 32) {
#if HAVE_TDM
    if (wave == 0) {
      if (k0 + 32 < KSKIP) {
        tdm_load_tile(W + k0 + 32,                          buf ? asOff0 : asOff1, 32, 64,  KSKIP);
        tdm_load_tile(H + (size_t)colBase * KSKIP + k0 + 32, buf ? hsOff0 : hsOff1, 32, 128, KSKIP);
        __builtin_amdgcn_s_wait_tensorcnt(2);
      } else {
        __builtin_amdgcn_s_wait_tensorcnt(0);
      }
    }
    __syncthreads();
#else
    { int r = tid >> 2, seg = tid & 3;
      *(uint4*)&as[0][r][seg * 8] = *(const uint4*)(W + (size_t)r * KSKIP + k0 + seg * 8);
      int c = tid >> 1, sg = tid & 1;
      const uint4* src = (const uint4*)(H + (size_t)(colBase + c) * KSKIP + k0 + sg * 16);
      *(uint4*)&hs[0][c][sg * 16]     = src[0];
      *(uint4*)&hs[0][c][sg * 16 + 8] = src[1];
      __syncthreads(); }
#endif

    FragBF af[2], bf[2];
#pragma unroll
    for (int ms = 0; ms < 2; ++ms) {
      int rA = wm * 32 + ms * 16 + n16;
      af[ms].q[0] = *(const uint4*)&as[buf][rA][8 * half];
      af[ms].q[1] = *(const uint4*)&as[buf][rA][16 + 8 * half];
    }
#pragma unroll
    for (int cs = 0; cs < 2; ++cs) {
      int cB = wc * 32 + cs * 16 + n16;
      bf[cs].q[0] = *(const uint4*)&hs[buf][cB][8 * half];
      bf[cs].q[1] = *(const uint4*)&hs[buf][cB][16 + 8 * half];
    }
#pragma unroll
    for (int ms = 0; ms < 2; ++ms)
#pragma unroll
      for (int cs = 0; cs < 2; ++cs)
        acc[ms][cs] = __builtin_amdgcn_wmma_f32_16x16x32_bf16(
            false, af[ms].v, false, bf[cs].v, (short)0, acc[ms][cs], false, false);
    __syncthreads();
#if HAVE_TDM
    buf ^= 1;
#endif
  }

#pragma unroll
  for (int ms = 0; ms < 2; ++ms)
#pragma unroll
    for (int cs = 0; cs < 2; ++cs) {
      int col = colBase + wc * 32 + cs * 16 + n16;
      int bb = col / Nn, nn = col % Nn;
      int v0 = wm * 32 + ms * 16 + 8 * half;
#pragma unroll
      for (int i = 0; i < 8; ++i)
        outS[(size_t)(bb * CSc + v0 + i) * Nn + nn] = acc[ms][cs][i] + bias[v0 + i];
    }
}

// ---------------- 1x1 graph-conv mix + residual + per-block partial sums ----------------
struct GcArgs {
  const unsigned short *h0, *h1a, *h2a, *h1b, *h2b;
  const float *w1, *b1, *w2, *b2, *x;
  float* out;       // xo written into xn region of d_out
  float* partials;  // [6656][2]
};

__global__ __launch_bounds__(256) void k_gc(GcArgs a) {
  __shared__ unsigned short ush[5][32][128];
  __shared__ float wsh[2][3072];
  __shared__ float red[256];
  int blk = blockIdx.x;
  int b = blk / 208, rem = blk % 208, l = rem >> 2, vt = rem & 3;
  int vBase = vt * 128;
  int tid = threadIdx.x;

  const unsigned short* Hs[5] = {a.h0, a.h1a, a.h2a, a.h1b, a.h2b};
  for (int d = 0; d < 5; ++d)
    for (int idx = tid; idx < 4096; idx += 256) {
      int c = idx >> 7, vv = idx & 127;
      ush[d][c][vv] = Hs[d][((size_t)(b * CCc + c) * Kk + l) * NPAD + vBase + vv];
    }
  for (int i = tid; i < 3072; i += 256) { wsh[0][i] = a.w1[i]; wsh[1][i] = a.w2[i]; }
  __syncthreads();

  int cr = tid >> 3, vg = tid & 7;
  float accv[16];
  float binit = a.b1[cr] + a.b2[cr];
#pragma unroll
  for (int vv = 0; vv < 16; ++vv) accv[vv] = binit;

  for (int c = 0; c < 32; ++c) {
    float w1a = wsh[0][cr * 96 + c], w1b = wsh[0][cr * 96 + 32 + c], w1c = wsh[0][cr * 96 + 64 + c];
    float w2a = wsh[1][cr * 96 + c], w2b = wsh[1][cr * 96 + 32 + c], w2c = wsh[1][cr * 96 + 64 + c];
    float w0 = w1a + w2a;
    int vb = vg * 16;
#pragma unroll
    for (int vv = 0; vv < 16; ++vv) {
      int vl = vb + vv;
      accv[vv] += w0 * bf2f(ush[0][c][vl]) + w1b * bf2f(ush[1][c][vl]) + w1c * bf2f(ush[2][c][vl])
                + w2b * bf2f(ush[3][c][vl]) + w2c * bf2f(ush[4][c][vl]);
    }
  }

  float s1 = 0.f, s2 = 0.f;
  for (int vv = 0; vv < 16; ++vv) {
    int v = vBase + vg * 16 + vv;
    if (v < Nn) {
      float xo = accv[vv] + a.x[(((size_t)b * CRc + cr) * Nn + v) * Tt + 12 + l];
      a.out[(((size_t)b * CRc + cr) * Nn + v) * Kk + l] = xo;
      s1 += xo; s2 += xo * xo;
    }
  }
  red[tid] = s1; __syncthreads();
  for (int st = 128; st > 0; st >>= 1) { if (tid < st) red[tid] += red[tid + st]; __syncthreads(); }
  if (tid == 0) a.partials[(size_t)blk * 2] = red[0];
  __syncthreads();
  red[tid] = s2; __syncthreads();
  for (int st = 128; st > 0; st >>= 1) { if (tid < st) red[tid] += red[tid + st]; __syncthreads(); }
  if (tid == 0) a.partials[(size_t)blk * 2 + 1] = red[0];
}

// ---------------- deterministic per-batch reduction of partials ----------------
__global__ __launch_bounds__(256) void k_reduce(const float* __restrict__ partials,
                                                float* __restrict__ sums) {
  __shared__ float r1[256], r2[256];
  int b = blockIdx.x, tid = threadIdx.x;
  float s1 = 0.f, s2 = 0.f;
  if (tid < 208) {
    s1 = partials[(size_t)(b * 208 + tid) * 2];
    s2 = partials[(size_t)(b * 208 + tid) * 2 + 1];
  }
  r1[tid] = s1; r2[tid] = s2; __syncthreads();
  for (int st = 128; st > 0; st >>= 1) {
    if (tid < st) { r1[tid] += r1[tid + st]; r2[tid] += r2[tid + st]; }
    __syncthreads();
  }
  if (tid == 0) { sums[b] = r1[0]; sums[32 + b] = r2[0]; }
}

// ---------------- layernorm finalize (in place on xn region) ----------------
__global__ __launch_bounds__(256) void k_final(float* __restrict__ out,
                                               const float* __restrict__ sums,
                                               const int* __restrict__ idx,
                                               const float* __restrict__ lnw,
                                               const float* __restrict__ lnb) {
  size_t i = (size_t)blockIdx.x * 256 + threadIdx.x;
  if (i >= XN_ELEMS) return;
  int l = (int)(i % Kk); size_t r = i / Kk;
  int v = (int)(r % Nn); r /= Nn;
  int cr = (int)(r % CRc); int b = (int)(r / CRc);
  const float cnt = (float)(CRc * Nn * Kk);   // 832000
  float mu  = sums[b] / cnt;
  float var = sums[32 + b] / cnt - mu * mu;
  float xo  = out[i];
  float xn  = (xo - mu) * rsqrtf(var + 1e-5f);
  int node = idx[v];
  size_t li = ((size_t)cr * Nn + node) * Kk + l;
  out[i] = xn * lnw[li] + lnb[li];
}

// ---------------- host launch ----------------
extern "C" void kernel_launch(void* const* d_in, const int* in_sizes, int n_in,
                              void* d_out, int out_size, void* d_ws, size_t ws_size,
                              hipStream_t stream) {
  (void)in_sizes; (void)n_in; (void)out_size; (void)ws_size;
  const float* x    = (const float*)d_in[0];
  const float* adp  = (const float*)d_in[1];
  const int*   idx  = (const int*)d_in[2];
  const float* fw[4] = {(const float*)d_in[3], (const float*)d_in[5], (const float*)d_in[7], (const float*)d_in[9]};
  const float* fb[4] = {(const float*)d_in[4], (const float*)d_in[6], (const float*)d_in[8], (const float*)d_in[10]};
  const float* gw[4] = {(const float*)d_in[11], (const float*)d_in[13], (const float*)d_in[15], (const float*)d_in[17]};
  const float* gb[4] = {(const float*)d_in[12], (const float*)d_in[14], (const float*)d_in[16], (const float*)d_in[18]};
  const float* skip_w = (const float*)d_in[19];
  const float* skip_b = (const float*)d_in[20];
  const float* gc1_w  = (const float*)d_in[21];
  const float* gc1_b  = (const float*)d_in[22];
  const float* gc2_w  = (const float*)d_in[23];
  const float* gc2_b  = (const float*)d_in[24];
  const float* lnw    = (const float*)d_in[25];
  const float* lnb    = (const float*)d_in[26];

  char* ws = (char*)d_ws;
  unsigned short* A1    = (unsigned short*)(ws + OFF_A1);
  unsigned short* A2    = (unsigned short*)(ws + OFF_A2);
  unsigned short* H0b   = (unsigned short*)(ws + OFF_H0);
  unsigned short* H1a   = (unsigned short*)(ws + OFF_H1A);
  unsigned short* H2a   = (unsigned short*)(ws + OFF_H2A);
  unsigned short* H1b   = (unsigned short*)(ws + OFF_H1B);
  unsigned short* H2b   = (unsigned short*)(ws + OFF_H2B);
  unsigned short* Hskip = (unsigned short*)(ws + OFF_HS);
  unsigned short* Wskip = (unsigned short*)(ws + OFF_WSK);
  float* partials = (float*)(ws + OFF_PART);
  float* sums     = (float*)(ws + OFF_SUMS);

  float* out  = (float*)d_out;
  float* outS = out + XN_ELEMS;

  k_adj<<<4, 256, 0, stream>>>(adp, A1, A2);
  k_cvt<<<(CSc * KSKIP + 255) / 256, 256, 0, stream>>>(skip_w, Wskip, CSc * KSKIP);
  k_padH<<<(COLS * 12) / 256, 256, 0, stream>>>(H0b);

  GateArgs ga;
  ga.x = x;
  for (int i = 0; i < 4; ++i) { ga.fw[i] = fw[i]; ga.fb[i] = fb[i]; ga.gw[i] = gw[i]; ga.gb[i] = gb[i]; }
  ga.H0 = H0b; ga.Hskip = Hskip;
  k_gate<<<Bb * Nn, 256, 0, stream>>>(ga);

  dim3 gp(NPAD / 64, COLS / 128);   // 8 x 416
  k_gemm_prop<<<gp, 256, 0, stream>>>(A1, H0b, H0b, H1a);
  k_gemm_prop<<<gp, 256, 0, stream>>>(A1, H1a, H0b, H2a);
  k_gemm_prop<<<gp, 256, 0, stream>>>(A2, H0b, H0b, H1b);
  k_gemm_prop<<<gp, 256, 0, stream>>>(A2, H1b, H0b, H2b);

  k_gemm_skip<<<SCOLS / 128, 256, 0, stream>>>(Wskip, Hskip, skip_b, outS);

  GcArgs gc;
  gc.h0 = H0b; gc.h1a = H1a; gc.h2a = H2a; gc.h1b = H1b; gc.h2b = H2b;
  gc.w1 = gc1_w; gc.b1 = gc1_b; gc.w2 = gc2_w; gc.b2 = gc2_b; gc.x = x;
  gc.out = out; gc.partials = partials;
  k_gc<<<Bb * Kk * 4, 256, 0, stream>>>(gc);          // 6656 blocks

  k_reduce<<<Bb, 256, 0, stream>>>(partials, sums);
  k_final<<<(unsigned)((XN_ELEMS + 255) / 256), 256, 0, stream>>>(out, sums, idx, lnw, lnb);
}
